// RecurrentDecoder_22746146800264
// MI455X (gfx1250) — matrix operand order repaired and checked by reading, not compiled, not run on hardware
//
#include <hip/hip_runtime.h>
#include <hip/hip_bf16.h>

// Problem constants
#define BB 8192   // batch
#define TT 100    // timesteps
#define II 30     // input size
#define HH 100    // hidden
#define GG 400    // 4*H gates
#define GP 404    // gate/xp LDS row stride (404 % 64 = 20 -> conflict-free halves)
#define KP 128    // K padded to multiple of 32 for wmma bf16
#define KS 136    // hS LDS row stride in ushorts (68 dwords = 4 mod 64 banks)
#define BT 32     // batch rows per block
#define NTILES 25 // GG/16

typedef __attribute__((ext_vector_type(16))) __bf16 v16bf;
typedef __attribute__((ext_vector_type(8)))  float  v8f;

union Frag {
    v16bf v;
    uint4 q[2];
};

__device__ __forceinline__ unsigned short f2bf(float f) {
    unsigned u = __float_as_uint(f);
    u += 0x7FFFu + ((u >> 16) & 1u);   // round-to-nearest-even
    return (unsigned short)(u >> 16);
}
__device__ __forceinline__ float bf2f(unsigned short s) {
    return __uint_as_float(((unsigned)s) << 16);
}
__device__ __forceinline__ float sigmoidf_fast(float x) {
    return 1.0f / (1.0f + __expf(-x));
}
__device__ __forceinline__ float tanhf_fast(float x) {
    float e = __expf(-2.0f * x);
    return (1.0f - e) / (1.0f + e);
}

// Convert W_hh [400][100] f32 -> bf16 [400][128] with zero K-padding.
__global__ void prep_whh_bf16(const float* __restrict__ W_hh,
                              unsigned short* __restrict__ WtB) {
    int idx = blockIdx.x * 256 + threadIdx.x;
    if (idx < GG * KP) {
        int n = idx / KP;
        int k = idx - n * KP;
        unsigned short v = 0;
        if (k < HH) v = f2bf(W_hh[n * HH + k]);
        WtB[idx] = v;
    }
}

__device__ __forceinline__ void load_bfrag(Frag& f, const unsigned short* WtB,
                                           int nt, int kt, int lmod, int lhalf) {
    // B layout (32x16 bf16): col n = lane%16; K = (lane/16)*8 .. +7 and
    // K = 16+(lane/16)*8 .. +7 (matches A fragment order).
    const uint4* p = (const uint4*)(WtB +
        (size_t)(nt * 16 + lmod) * KP + kt * 32 + lhalf * 8);
    f.q[0] = p[0];
    f.q[1] = p[2];
}

__global__ __launch_bounds__(256)
void lstm_decoder_fused(const float* __restrict__ x,
                        const float* __restrict__ W_ih,
                        const float* __restrict__ b_ih,
                        const float* __restrict__ b_hh,
                        const unsigned short* __restrict__ WtB, // [GG][KP] bf16
                        const float* __restrict__ W_d,
                        const float* __restrict__ b_d,
                        float* __restrict__ out) {
    __shared__ float          xpS[BT][GP];    // ~50.6 KB: x@W_ih^T + biases
    __shared__ float          gateS[BT][GP];  // ~50.6 KB: per-step gates
    __shared__ float          cS[BT][HH];     // 12.5 KB
    __shared__ unsigned short hS[BT][KS];     // 8.5 KB (bf16, padded stride)
    __shared__ float          wdS[HH];

    const int tid   = threadIdx.x;
    const int wave  = tid >> 5;
    const int lane  = tid & 31;
    const int lhalf = lane >> 4;    // 0 or 1
    const int lmod  = lane & 15;
    const int b0    = blockIdx.x * BT;

    // ---- Persistent B-fragments of W_hh in registers --------------------
    // Waves 0..7 own ntiles {w, w+8, w+16}; wave 0 additionally owns nt=24.
    Frag Bf[3][4];
    Frag Bf24[4];
#pragma unroll
    for (int i = 0; i < 3; ++i)
#pragma unroll
        for (int kt = 0; kt < 4; ++kt)
            load_bfrag(Bf[i][kt], WtB, wave + 8 * i, kt, lmod, lhalf);
    if (wave == 0)
#pragma unroll
        for (int kt = 0; kt < 4; ++kt)
            load_bfrag(Bf24[kt], WtB, 24, kt, lmod, lhalf);

    // ---- Init LDS state -------------------------------------------------
    for (int e = tid; e < BT * KS; e += 256) (&hS[0][0])[e] = 0;
    for (int e = tid; e < BT * HH; e += 256) (&cS[0][0])[e] = 0.0f;
    for (int e = tid; e < HH;     e += 256) wdS[e] = W_d[e];

    // xp = x @ W_ih^T + b_ih + b_hh  (tiny K=30; done once per block)
    for (int e = tid; e < BT * GG; e += 256) {
        int b = e / GG, n = e - b * GG;
        float s = b_ih[n] + b_hh[n];
        const float* xr = x    + (size_t)(b0 + b) * II;
        const float* wr = W_ih + (size_t)n * II;
#pragma unroll
        for (int k = 0; k < II; ++k) s += xr[k] * wr[k];
        xpS[b][n] = s;
    }
    __syncthreads();

    const float bd = b_d[0];

    // ---- Recurrence over T ---------------------------------------------
    for (int t = 0; t < TT; ++t) {
        // Phase 1: gates = xp + h @ W_hh^T  via WMMA bf16
#pragma unroll
        for (int m = 0; m < BT / 16; ++m) {
            // A layout (16x32 bf16): row = lane%16; VGPR0-3 hold
            // K = (lane/16)*8 .. +7, VGPR4-7 hold K = 16+(lane/16)*8 .. +7
            Frag Af[4];
#pragma unroll
            for (int kt = 0; kt < 4; ++kt) {
                const uint4* p = (const uint4*)(&hS[m * 16 + lmod][kt * 32 + lhalf * 8]);
                Af[kt].q[0] = p[0];   // k0 .. k0+7
                Af[kt].q[1] = p[2];   // k0+16 .. k0+23
            }

            // C layout: VGPR r, lane l -> m = r + 8*(l/16), n = l%16
            const int rowb = m * 16 + 8 * lhalf;

#pragma unroll
            for (int i = 0; i < 3; ++i) {   // unconditional: nt <= 23 < 25
                const int nt = wave + 8 * i;
                v8f acc;
#pragma unroll
                for (int r = 0; r < 8; ++r)
                    acc[r] = xpS[rowb + r][nt * 16 + lmod];
#pragma unroll
                for (int kt = 0; kt < 4; ++kt)
                    acc = __builtin_amdgcn_wmma_f32_16x16x32_bf16(
                        false, Af[kt].v, false, Bf[i][kt].v,
                        (short)0, acc, false, false);
#pragma unroll
                for (int r = 0; r < 8; ++r)
                    gateS[rowb + r][nt * 16 + lmod] = acc[r];
            }
            if (wave == 0) {                // tail tile nt = 24
                v8f acc;
#pragma unroll
                for (int r = 0; r < 8; ++r)
                    acc[r] = xpS[rowb + r][24 * 16 + lmod];
#pragma unroll
                for (int kt = 0; kt < 4; ++kt)
                    acc = __builtin_amdgcn_wmma_f32_16x16x32_bf16(
                        false, Af[kt].v, false, Bf24[kt].v,
                        (short)0, acc, false, false);
#pragma unroll
                for (int r = 0; r < 8; ++r)
                    gateS[rowb + r][24 * 16 + lmod] = acc[r];
            }
        }
        __syncthreads();

        // Phase 2: elementwise LSTM cell update (i,f,g,o -> c,h)
        for (int e = tid; e < BT * HH; e += 256) {
            int b = e / HH, j = e - b * HH;
            float gi = gateS[b][j];
            float gf = gateS[b][HH + j];
            float gg = gateS[b][2 * HH + j];
            float go = gateS[b][3 * HH + j];
            float iv = sigmoidf_fast(gi);
            float fv = sigmoidf_fast(gf);
            float gv = tanhf_fast(gg);
            float ov = sigmoidf_fast(go);
            float c  = fv * cS[b][j] + iv * gv;
            cS[b][j] = c;
            float h  = ov * tanhf_fast(c);
            hS[b][j] = f2bf(h);
        }
        __syncthreads();

        // Phase 3: out[b][t] = h . W_d + b_d  (8 lanes per batch row)
        {
            int b = tid >> 3, sl = tid & 7;
            float s = 0.0f;
#pragma unroll
            for (int jj = 0; jj < 13; ++jj) {
                int j = sl + (jj << 3);
                if (j < HH) s += bf2f(hS[b][j]) * wdS[j];
            }
            s += __shfl_xor(s, 1);
            s += __shfl_xor(s, 2);
            s += __shfl_xor(s, 4);
            if (sl == 0) out[(size_t)(b0 + b) * TT + t] = s + bd;
        }
        // No extra barrier: next-step phase 1 only reads hS/xpS and writes
        // gateS; the barrier after phase 1 orders it against phase 2's reads.
    }
}

extern "C" void kernel_launch(void* const* d_in, const int* in_sizes, int n_in,
                              void* d_out, int out_size, void* d_ws, size_t ws_size,
                              hipStream_t stream) {
    const float* x    = (const float*)d_in[0];
    const float* W_ih = (const float*)d_in[1];
    const float* W_hh = (const float*)d_in[2];
    const float* b_ih = (const float*)d_in[3];
    const float* b_hh = (const float*)d_in[4];
    const float* W_d  = (const float*)d_in[5];
    const float* b_d  = (const float*)d_in[6];
    float* out = (float*)d_out;

    unsigned short* WtB = (unsigned short*)d_ws;  // 400*128*2 = 100 KB

    prep_whh_bf16<<<(GG * KP + 255) / 256, 256, 0, stream>>>(W_hh, WtB);
    lstm_decoder_fused<<<BB / BT, 256, 0, stream>>>(
        x, W_ih, b_ih, b_hh, WtB, W_d, b_d, out);
}
